// _Fp32AccumRef_74612171866698
// MI455X (gfx1250) — matrix operand order repaired and checked by reading, not compiled, not run on hardware
//
#include <hip/hip_runtime.h>
#include <hip/hip_bf16.h>
#include <math.h>

// Problem constants (from reference)
#define BB 8
#define TT 2048
#define DIN 512
#define DD 1024
#define MM (BB * TT)          // 16384 rows
#define LN_EPS 1e-5f

typedef __attribute__((ext_vector_type(16))) _Float16 v16h;
typedef __attribute__((ext_vector_type(8)))  _Float16 v8h;
typedef __attribute__((ext_vector_type(8)))  float    v8f;

#define LDA 40   // halves per A row in LDS (80B rows, 16B aligned)
#define LDB 48   // halves per B row in LDS (96B rows, 16B aligned)

// A fragment, 16x32 f16: lane<16 holds M=lane, K=0..15;
// lane>=16 holds M=lane-16, K=16..31.
static __device__ __forceinline__ v16h load_a_frag(const _Float16* As, int mRow, int lane) {
    const int k0 = (lane >= 16) ? 16 : 0;
    const v8h lo = *(const v8h*)&As[mRow * LDA + k0];
    const v8h hi = *(const v8h*)&As[mRow * LDA + k0 + 8];
    v16h r;
#pragma unroll
    for (int i = 0; i < 8; ++i) { r[i] = lo[i]; r[i + 8] = hi[i]; }
    return r;
}

// B fragment, 32x16 f16: lane<16 holds N=lane, K=0..15; lane>=16 holds N=lane-16,
// K=16..31. LDS holds W transposed as [n][k], so each lane reads 16 contiguous halves.
static __device__ __forceinline__ v16h load_b_frag(const _Float16* Bs, int nRow, int lane) {
    const int k0 = (lane >= 16) ? 16 : 0;
    const v8h lo = *(const v8h*)&Bs[nRow * LDB + k0];
    const v8h hi = *(const v8h*)&Bs[nRow * LDB + k0 + 8];
    v16h r;
#pragma unroll
    for (int i = 0; i < 8; ++i) { r[i] = lo[i]; r[i + 8] = hi[i]; }
    return r;
}

// Stage one K-step of tiles (x: 128x32, Wd/Wb: 32x128 transposed) f32 -> f16 into LDS.
// Phase 1 issues ALL 12 global_load_b128 into distinct registers (one wait total),
// phase 2 converts and stores to LDS — avoids per-load s_wait_loadcnt serialization.
static __device__ __forceinline__ void stage_tiles(
    _Float16* As, _Float16* BsD, _Float16* BsB,
    const float* __restrict__ x, const float* __restrict__ Wd, const float* __restrict__ Wb,
    int mBase, int nBase, int kt, int tid)
{
    const int r  = tid >> 3;        // 0..31
    const int c4 = (tid & 7) * 4;   // 0..28
    const int n4 = (tid & 31) * 4;  // 0..124
    const int k  = tid >> 5;        // 0..7

    float4 fx[4], fd[4], fb[4];
    // ---- phase 1: issue all loads ----
#pragma unroll
    for (int p = 0; p < 4; ++p)
        fx[p] = *(const float4*)&x[(size_t)(mBase + r + p * 32) * DIN + kt + c4];
#pragma unroll
    for (int p = 0; p < 4; ++p) {
        const size_t gro = (size_t)(kt + k + p * 8) * DD + nBase + n4;
        fd[p] = *(const float4*)&Wd[gro];
        fb[p] = *(const float4*)&Wb[gro];
    }
    // ---- phase 2: convert + LDS store ----
#pragma unroll
    for (int p = 0; p < 4; ++p) {
        _Float16* dst = &As[(r + p * 32) * LDA + c4];
        dst[0] = (_Float16)fx[p].x; dst[1] = (_Float16)fx[p].y;
        dst[2] = (_Float16)fx[p].z; dst[3] = (_Float16)fx[p].w;
    }
#pragma unroll
    for (int p = 0; p < 4; ++p) {
        const int kk = k + p * 8;
        BsD[(n4 + 0) * LDB + kk] = (_Float16)fd[p].x;
        BsD[(n4 + 1) * LDB + kk] = (_Float16)fd[p].y;
        BsD[(n4 + 2) * LDB + kk] = (_Float16)fd[p].z;
        BsD[(n4 + 3) * LDB + kk] = (_Float16)fd[p].w;
        BsB[(n4 + 0) * LDB + kk] = (_Float16)fb[p].x;
        BsB[(n4 + 1) * LDB + kk] = (_Float16)fb[p].y;
        BsB[(n4 + 2) * LDB + kk] = (_Float16)fb[p].z;
        BsB[(n4 + 3) * LDB + kk] = (_Float16)fb[p].w;
    }
}

// Fused dual-GEMM + discretization epilogue, double-buffered LDS.
// dlin = x@Wd + bd, blin = x@Wb + bb per 128x128 tile via WMMA, then
// a_step = exp(-softplus(dlin)*exp(A_log)), b_step = softplus(dlin)*blin.
__global__ void __launch_bounds__(256)
snn_gemm_disc_kernel(const float* __restrict__ x,
                     const float* __restrict__ Wd, const float* __restrict__ bd,
                     const float* __restrict__ Wb, const float* __restrict__ bb,
                     const float* __restrict__ A_log,
                     float* __restrict__ a_ws, float* __restrict__ b_ws) {
    __shared__ __align__(16) _Float16 AsBuf [2][128 * LDA];
    __shared__ __align__(16) _Float16 BsDBuf[2][128 * LDB];
    __shared__ __align__(16) _Float16 BsBBuf[2][128 * LDB];

    const int tid   = threadIdx.x;
    const int lane  = tid & 31;
    const int wave  = tid >> 5;      // 0..7
    const int waveM = wave & 3;      // 4 waves along M -> 32 rows each
    const int waveN = wave >> 2;     // 2 waves along N -> 64 cols each
    const int mBase = blockIdx.y * 128;
    const int nBase = blockIdx.x * 128;

    v8f accD[2][4];
    v8f accB[2][4];
#pragma unroll
    for (int mi = 0; mi < 2; ++mi)
#pragma unroll
        for (int ni = 0; ni < 4; ++ni) { accD[mi][ni] = (v8f)(0.0f); accB[mi][ni] = (v8f)(0.0f); }

    // Prologue: stage first tile into buffer 0.
    stage_tiles(AsBuf[0], BsDBuf[0], BsBBuf[0], x, Wd, Wb, mBase, nBase, 0, tid);
    __syncthreads();

    int p = 0;
    for (int kt = 0; kt < DIN; kt += 32) {
        // Prefetch next tile into the other buffer while this tile's WMMAs run.
        if (kt + 32 < DIN)
            stage_tiles(AsBuf[p ^ 1], BsDBuf[p ^ 1], BsBBuf[p ^ 1],
                        x, Wd, Wb, mBase, nBase, kt + 32, tid);

        const _Float16* Asc  = AsBuf[p];
        const _Float16* BsD_ = BsDBuf[p];
        const _Float16* BsB_ = BsBBuf[p];

        // ---- WMMA: 2(M) x 4(N) subtiles x 2 weight matrices ----
#pragma unroll
        for (int mi = 0; mi < 2; ++mi) {
            const v16h af = load_a_frag(Asc, waveM * 32 + mi * 16 + (lane & 15), lane);
#pragma unroll
            for (int ni = 0; ni < 4; ++ni) {
                const int nRow = waveN * 64 + ni * 16 + (lane & 15);
                const v16h bfd = load_b_frag(BsD_, nRow, lane);
                accD[mi][ni] = __builtin_amdgcn_wmma_f32_16x16x32_f16(
                    false, af, false, bfd, (short)0, accD[mi][ni], false, false);
                const v16h bfb = load_b_frag(BsB_, nRow, lane);
                accB[mi][ni] = __builtin_amdgcn_wmma_f32_16x16x32_f16(
                    false, af, false, bfb, (short)0, accB[mi][ni], false, false);
            }
        }
        __syncthreads();   // next tile fully staged AND this tile's reads done
        p ^= 1;
    }

    // ---- epilogue: bias + softplus + exp discretization, write a/b streams ----
    // C/D layout: lane<16 -> N=lane, rows M=r; lane>=16 -> N=lane-16, rows M=r+8.
    const int lane15 = lane & 15;
    const int laneHi = (lane >> 4) & 1;
#pragma unroll
    for (int mi = 0; mi < 2; ++mi) {
#pragma unroll
        for (int ni = 0; ni < 4; ++ni) {
            const int n = nBase + waveN * 64 + ni * 16 + lane15;
            const float bdv = bd[n];
            const float bbv = bb[n];
            const float Av  = __expf(A_log[n]);
#pragma unroll
            for (int r = 0; r < 8; ++r) {
                const int m = mBase + waveM * 32 + mi * 16 + r + laneHi * 8;
                const float vd = accD[mi][ni][r] + bdv;
                // softplus with overflow guard; fast v_exp/v_log path
                const float delta = (vd > 20.f) ? vd : __logf(1.f + __expf(vd));
                const float vb = accB[mi][ni][r] + bbv;
                const size_t off = (size_t)m * DD + n;
                a_ws[off] = __expf(-delta * Av);
                b_ws[off] = delta * vb;
            }
        }
    }
}

// Sequential spiking scan over T; one thread per (b,d). Coalesced over d.
// Latency-bound (8192 independent chains): software-prefetch 8 timesteps ahead
// (global_prefetch_b8) so each wave keeps many HBM lines in flight.
__global__ void __launch_bounds__(256)
snn_scan_kernel(const float* __restrict__ a_ws, const float* __restrict__ b_ws,
                const float* __restrict__ threshold,
                float* __restrict__ h_out, float* __restrict__ s_out) {
    const int id = blockIdx.x * blockDim.x + threadIdx.x;   // 0..8191
    const int b = id >> 10;
    const int d = id & (DD - 1);
    const float thr = threshold[d];
    float h = 0.f;
    const size_t base = (size_t)b * TT * DD + d;
#pragma unroll 8
    for (int t = 0; t < TT; ++t) {
        const size_t off = base + (size_t)t * DD;
        if (t + 8 < TT) {
            __builtin_prefetch(&a_ws[off + 8 * DD], 0, 1);
            __builtin_prefetch(&b_ws[off + 8 * DD], 0, 1);
        }
        const float a  = a_ws[off];
        const float bv = b_ws[off];
        h = fmaf(a, h, bv);
        const float s = (h - thr > 0.f) ? 1.f : 0.f;
        h = (s > 0.f) ? 0.f : h;
        h_out[off] = h;
        s_out[off] = s;
    }
}

// LayerNorm over D=1024 per row, in place on d_out[0 .. M*D). One block per row.
__global__ void __launch_bounds__(256)
snn_ln_kernel(float* __restrict__ out,
              const float* __restrict__ gamma, const float* __restrict__ beta) {
    __shared__ float ssum[8];
    __shared__ float ssq[8];
    const int row  = blockIdx.x;
    const int tid  = threadIdx.x;
    const int lane = tid & 31;
    const int wid  = tid >> 5;
    float* rowp = out + (size_t)row * DD;

    const float4 v = ((const float4*)rowp)[tid];       // 256 * 4 = 1024
    float s  = v.x + v.y + v.z + v.w;
    float s2 = v.x * v.x + v.y * v.y + v.z * v.z + v.w * v.w;
#pragma unroll
    for (int off = 16; off > 0; off >>= 1) {
        s  += __shfl_xor(s,  off, 32);
        s2 += __shfl_xor(s2, off, 32);
    }
    if (lane == 0) { ssum[wid] = s; ssq[wid] = s2; }
    __syncthreads();
    float tot = 0.f, tot2 = 0.f;
#pragma unroll
    for (int i = 0; i < 8; ++i) { tot += ssum[i]; tot2 += ssq[i]; }
    const float mu  = tot * (1.0f / DD);
    const float var = tot2 * (1.0f / DD) - mu * mu;
    const float inv = rsqrtf(var + LN_EPS);

    const float4 g  = ((const float4*)gamma)[tid];
    const float4 be = ((const float4*)beta)[tid];
    float4 o;
    o.x = (v.x - mu) * inv * g.x + be.x;
    o.y = (v.y - mu) * inv * g.y + be.y;
    o.z = (v.z - mu) * inv * g.z + be.z;
    o.w = (v.w - mu) * inv * g.w + be.w;
    ((float4*)rowp)[tid] = o;
}

extern "C" void kernel_launch(void* const* d_in, const int* in_sizes, int n_in,
                              void* d_out, int out_size, void* d_ws, size_t ws_size,
                              hipStream_t stream) {
    const float* x     = (const float*)d_in[0];
    const float* Wd    = (const float*)d_in[1];
    const float* bd    = (const float*)d_in[2];
    const float* Wb    = (const float*)d_in[3];
    const float* bb    = (const float*)d_in[4];
    const float* A_log = (const float*)d_in[5];
    const float* thr   = (const float*)d_in[6];
    const float* gamma = (const float*)d_in[7];
    const float* beta  = (const float*)d_in[8];

    const size_t nElem = (size_t)MM * DD;   // 16,777,216
    float* out    = (float*)d_out;          // [B,T,D] normalized h
    float* spikes = out + nElem;            // [B,T,D] spikes
    float* a_ws   = (float*)d_ws;           // [M,D] a_step
    float* b_ws   = a_ws + nElem;           // [M,D] b_step  (needs 128MB ws)

    dim3 gGemm(DD / 128, MM / 128);         // (8, 128)
    snn_gemm_disc_kernel<<<gGemm, 256, 0, stream>>>(x, Wd, bd, Wb, bb, A_log, a_ws, b_ws);

    snn_scan_kernel<<<(BB * DD) / 256, 256, 0, stream>>>(a_ws, b_ws, thr, out, spikes);

    snn_ln_kernel<<<MM, 256, 0, stream>>>(out, gamma, beta);
}
